// GatedDIP_5016521802464
// MI455X (gfx1250) — matrix operand order repaired and checked by reading, not compiled, not run on hardware
//
#include <hip/hip_runtime.h>
#include <hip/hip_fp16.h>
#include <math.h>

typedef __attribute__((ext_vector_type(16))) _Float16 v16h;
typedef __attribute__((ext_vector_type(8)))  _Float16 v8h;
typedef __attribute__((ext_vector_type(8)))  float    v8f;
typedef __attribute__((ext_vector_type(4)))  int      v4i;
typedef __attribute__((ext_vector_type(4)))  unsigned v4u;
typedef __attribute__((ext_vector_type(8)))  int      v8i;

union V16H { v16h v; v8h h[2]; };

#define DEV __device__ __forceinline__

// ---------------------------------------------------------------- gfx1250 async global->LDS
#if defined(__has_builtin)
#if __has_builtin(__builtin_amdgcn_global_load_async_to_lds_b128)
#define HAVE_ASYNC_LDS 1
#endif
#if __has_builtin(__builtin_amdgcn_tensor_load_to_lds) && __has_builtin(__builtin_amdgcn_s_wait_tensorcnt)
#define HAVE_TDM 1
#endif
#endif

#ifdef HAVE_ASYNC_LDS
DEV void async_b128(const void* gsrc, void* ldst) {
    __builtin_amdgcn_global_load_async_to_lds_b128((v4i*)gsrc, (v4i*)ldst, 0, 0);
}
#endif

DEV void wait_async0() {
#if defined(__has_builtin)
#if __has_builtin(__builtin_amdgcn_s_wait_asynccnt)
    __builtin_amdgcn_s_wait_asynccnt(0);
#else
    asm volatile("s_wait_asynccnt 0x0" ::: "memory");
#endif
#else
    asm volatile("s_wait_asynccnt 0x0" ::: "memory");
#endif
}

#ifdef HAVE_TDM
// 2D tile load via Tensor Data Mover: tile (rows x 64 halves) from [dim1][dim0] f16 matrix
DEV void tdm_load_2d_f16(const _Float16* gsrc, void* lds, int dim0, int dim1,
                         int tile0, int tile1) {
    unsigned lds_off = (unsigned)(size_t)lds;                  // LDS aperture: low 32 bits = LDS offset
    unsigned long long ga = (unsigned long long)(size_t)gsrc;  // tile start
    v4u g0 = { 1u,                                             // count=1 valid descriptor
               lds_off,
               (unsigned)ga,
               (unsigned)((ga >> 32) & 0x01FFFFFFu) | (2u << 30) };  // addr[56:32] | type=2
    v8i g1;
    g1[0] = 0x00010000;                                        // data_size=2B, no multicast
    g1[1] = (int)(((unsigned)dim0 & 0xFFFFu) << 16);           // tensor_dim0 lo16 -> bits 63:48
    g1[2] = (int)((((unsigned)dim0 >> 16) & 0xFFFFu) |
                  (((unsigned)dim1 & 0xFFFFu) << 16));          // dim0 hi16 | tensor_dim1 lo16
    g1[3] = (int)((((unsigned)dim1 >> 16) & 0xFFFFu) |
                  (((unsigned)tile0 & 0xFFFFu) << 16));         // dim1 hi16 | tile_dim0
    g1[4] = tile1 & 0xFFFF;                                    // tile_dim1 (tile_dim2=0)
    g1[5] = dim0;                                              // tensor_dim0_stride lo32
    g1[6] = 0;                                                 // stride hi | dim1_stride lo
    g1[7] = 0;
    v4i gz = { 0, 0, 0, 0 };
    v8i gz8 = { 0, 0, 0, 0, 0, 0, 0, 0 };
    __builtin_amdgcn_tensor_load_to_lds(g0, g1, gz, gz, gz8, 0);
}
#endif

// ---------------------------------------------------------------- helpers
DEV float tanh01f(float x) { return tanhf(x) * 0.5f + 0.5f; }
DEV float trangef(float x, float l, float r) { return tanh01f(x) * (r - l) + l; }
DEV unsigned f2ord(float f) {
    unsigned u = __float_as_uint(f);
    return (u & 0x80000000u) ? ~u : (u | 0x80000000u);
}
DEV float ord2f(unsigned e) {
    unsigned u = (e & 0x80000000u) ? (e ^ 0x80000000u) : ~e;
    return __uint_as_float(u);
}

// ---------------------------------------------------------------- convert
__global__ void f32_to_f16_k(const float* __restrict__ src, _Float16* __restrict__ dst, long n) {
    long i = blockIdx.x * (long)blockDim.x + threadIdx.x;
    if (i < n) dst[i] = (_Float16)src[i];
}

// ---------------------------------------------------------------- conv 3x3 SAME, implicit GEMM via WMMA
// M = Cout, N = 8*H*W, K = Cin*9.  Tile 64(M) x 64(N) x 64(K); 8 waves; 4 WMMA/wave/iter.
__global__ __launch_bounds__(256)
void conv3x3_wmma_k(const _Float16* __restrict__ in,   // [8][Cin][H][W]
                    const _Float16* __restrict__ wgt,  // [Cout][Cin*9]
                    const float*    __restrict__ bias, // [Cout]
                    _Float16*       __restrict__ out,  // [8][Cout][H][W]
                    int Cin, int Cout, int H, int W)
{
    __shared__ alignas(16) _Float16 Asr[64][64];
    __shared__ alignas(16) _Float16 Bsr[64][64];
    const int HW   = H * W;
    const int Ntot = 8 * HW;
    const int K    = Cin * 9;
    const bool fastK = (K & 63) == 0;          // true for all layers except L0 (K=27)
    const int n0   = blockIdx.x * 64;
    const int m0   = blockIdx.y * 64;
    const int t    = threadIdx.x;
    const int lane = t & 31;
    const int wv   = t >> 5;
    const int mB   = (wv >> 1) * 16;           // wave's M sub-tile
    const int nB   = (wv & 1) * 32;            // wave's N sub-tile (2 frags)

    v8f c0 = {}; v8f c1 = {};

    // staging coordinates: each thread fills 16 halves of A and 16 of B
    const int ar  = t >> 2;                    // tile row 0..63
    const int ac0 = (t & 3) * 16;              // tile col start (k)
    const int gm  = m0 + ar;
    const int gn  = n0 + ar;
    const bool nok = gn < Ntot;
    int bb = 0, by = 0, bx = 0;
    { int g = nok ? gn : 0; bb = g / HW; int r = g % HW; by = r / W; bx = r % W; }

    for (int k0 = 0; k0 < K; k0 += 64) {
        // ---- stage A (weights) ----
        if (fastK) {
            __builtin_prefetch(&wgt[(long)gm * K + k0 + 64], 0, 1);   // global_prefetch_b8
#ifdef HAVE_TDM
            if (t == 0)
                tdm_load_2d_f16(&wgt[(long)m0 * K + k0], &Asr[0][0], K, Cout, 64, 64);
#elif defined(HAVE_ASYNC_LDS)
            const _Float16* src = &wgt[(long)gm * K + k0 + ac0];
            async_b128(src,     &Asr[ar][ac0]);
            async_b128(src + 8, &Asr[ar][ac0 + 8]);
#else
            const _Float16* src = &wgt[(long)gm * K + k0 + ac0];
            *(v8h*)&Asr[ar][ac0]     = *(const v8h*)src;
            *(v8h*)&Asr[ar][ac0 + 8] = *(const v8h*)(src + 8);
#endif
        } else {
            v8h a0, a1;
            #pragma unroll
            for (int j = 0; j < 8; ++j) {
                int gk0 = k0 + ac0 + j, gk1 = gk0 + 8;
                _Float16 t0 = wgt[(long)gm * K + min(gk0, K - 1)];
                _Float16 t1 = wgt[(long)gm * K + min(gk1, K - 1)];
                a0[j] = (gk0 < K) ? t0 : (_Float16)0.f;
                a1[j] = (gk1 < K) ? t1 : (_Float16)0.f;
            }
            *(v8h*)&Asr[ar][ac0]     = a0;
            *(v8h*)&Asr[ar][ac0 + 8] = a1;
        }
        // ---- stage B (im2col gather, unconditional clamped loads + select) ----
        {
            v8h b0, b1;
            #pragma unroll
            for (int j = 0; j < 16; ++j) {
                int gk  = k0 + ac0 + j;
                int gkc = min(gk, K - 1);
                int cin = gkc / 9; int r9 = gkc - cin * 9;
                int kh  = r9 / 3;  int kw = r9 - kh * 3;
                int yy = by + kh - 1, xx = bx + kw - 1;
                bool valid = nok && (gk < K) && (yy >= 0) && (yy < H) && (xx >= 0) && (xx < W);
                int yyc = min(max(yy, 0), H - 1), xxc = min(max(xx, 0), W - 1);
                _Float16 tv = in[(((long)bb * Cin + cin) * H + yyc) * W + xxc];
                _Float16 rv = valid ? tv : (_Float16)0.f;
                if (j < 8) b0[j] = rv; else b1[j - 8] = rv;
            }
            *(v8h*)&Bsr[ar][ac0]     = b0;
            *(v8h*)&Bsr[ar][ac0 + 8] = b1;
        }
#ifdef HAVE_TDM
        if (fastK && wv == 0) __builtin_amdgcn_s_wait_tensorcnt(0);
#elif defined(HAVE_ASYNC_LDS)
        if (fastK) wait_async0();
#endif
        __syncthreads();
        // ---- fragments + 4 WMMAs ----
        const int kh = (lane < 16) ? 0 : 8;    // A half-offset
        const int kb = (lane < 16) ? 0 : 8;    // B half-offset
        const int arow = mB + (lane & 15);
        const int nrow = nB + (lane & 15);
        V16H A0, A1, B00, B01, B10, B11;
        A0.h[0]  = *(const v8h*)&Asr[arow][kh];       A0.h[1]  = *(const v8h*)&Asr[arow][kh + 16];
        A1.h[0]  = *(const v8h*)&Asr[arow][32 + kh];  A1.h[1]  = *(const v8h*)&Asr[arow][32 + kh + 16];
        B00.h[0] = *(const v8h*)&Bsr[nrow][kb];       B00.h[1] = *(const v8h*)&Bsr[nrow][kb + 16];
        B01.h[0] = *(const v8h*)&Bsr[nrow][32 + kb];  B01.h[1] = *(const v8h*)&Bsr[nrow][32 + kb + 16];
        B10.h[0] = *(const v8h*)&Bsr[nrow + 16][kb];      B10.h[1] = *(const v8h*)&Bsr[nrow + 16][kb + 16];
        B11.h[0] = *(const v8h*)&Bsr[nrow + 16][32 + kb]; B11.h[1] = *(const v8h*)&Bsr[nrow + 16][32 + kb + 16];
        c0 = __builtin_amdgcn_wmma_f32_16x16x32_f16(false, A0.v, false, B00.v, (short)0, c0, false, false);
        c1 = __builtin_amdgcn_wmma_f32_16x16x32_f16(false, A0.v, false, B10.v, (short)0, c1, false, false);
        c0 = __builtin_amdgcn_wmma_f32_16x16x32_f16(false, A1.v, false, B01.v, (short)0, c0, false, false);
        c1 = __builtin_amdgcn_wmma_f32_16x16x32_f16(false, A1.v, false, B11.v, (short)0, c1, false, false);
        __syncthreads();
    }
    // ---- epilogue: bias + ReLU, scatter to NCHW ----
    #pragma unroll
    for (int r = 0; r < 8; ++r) {
        int mrow = m0 + mB + ((lane < 16) ? r : r + 8);
        if (mrow >= Cout) continue;
        float bsv = bias[mrow];
        #pragma unroll
        for (int f = 0; f < 2; ++f) {
            int ncol = n0 + nB + f * 16 + (lane & 15);
            if (ncol >= Ntot) continue;
            float v = (f == 0 ? c0[r] : c1[r]) + bsv;
            v = v > 0.f ? v : 0.f;
            int b2 = ncol / HW; int rr = ncol % HW; int yy = rr / W; int xx = rr % W;
            out[(((long)b2 * Cout + mrow) * H + yy) * W + xx] = (_Float16)v;
        }
    }
}

// ---------------------------------------------------------------- 2x2 maxpool (f16)
__global__ void maxpool2_k(const _Float16* __restrict__ in, _Float16* __restrict__ out,
                           int C, int H, int W)
{
    int Ho = H >> 1, Wo = W >> 1;
    long n = (long)8 * C * Ho * Wo;
    long i = blockIdx.x * (long)blockDim.x + threadIdx.x;
    if (i >= n) return;
    int xo = (int)(i % Wo); long r = i / Wo;
    int yo = (int)(r % Ho); r /= Ho;
    int c  = (int)(r % C);  int b = (int)(r / C);
    const _Float16* p = in + (((long)b * C + c) * H + yo * 2) * W + xo * 2;
    float a0 = (float)p[0], a1 = (float)p[1], a2 = (float)p[W], a3 = (float)p[W + 1];
    out[i] = (_Float16)fmaxf(fmaxf(a0, a1), fmaxf(a2, a3));
}

// ---------------------------------------------------------------- FC GEMM via WMMA (M<=16)
// requires N % 128 == 0 (exact grid: N=4096,256), K % 32 == 0 (25088,4096).
// A rows >= M may contain garbage: they only affect D rows >= M which are never stored,
// and the over-read stays inside the workspace buffers.
__global__ __launch_bounds__(256)
void gemm_wmma_k(const _Float16* __restrict__ Am,  // [M][K]
                 const _Float16* __restrict__ Bm,  // [K][N]
                 const float*    __restrict__ bias,
                 _Float16*       __restrict__ Cm,  // [M][N]
                 int M, int N, int K, int relu)
{
    __shared__ alignas(16) _Float16 Asr[16][32];
    __shared__ alignas(16) _Float16 Bsr[128][32];
    const int n0 = blockIdx.x * 128;
    const int t = threadIdx.x, lane = t & 31, wv = t >> 5;
    v8f acc = {};
    for (int k0 = 0; k0 < K; k0 += 32) {
        // stage A: 512 halves, 16B per thread for t<64 (async global->LDS when available)
#ifdef HAVE_ASYNC_LDS
        if (t < 64) {
            int arr = t >> 2, ac = (t & 3) * 8;
            async_b128(&Am[(long)arr * K + k0 + ac], &Asr[arr][ac]);
        }
#else
        {
            int idx = t * 2;
            int arr = idx >> 5, ac = idx & 31;
            const _Float16* src = &Am[(long)min(arr, M - 1) * K + k0 + ac];
            _Float16 v0 = src[0], v1 = src[1];
            bool vr = arr < M;
            Asr[arr][ac]     = vr ? v0 : (_Float16)0.f;
            Asr[arr][ac + 1] = vr ? v1 : (_Float16)0.f;
        }
#endif
        { // stage B: coalesced v8h loads along N, transpose into [n][k]
            int kr = t >> 3;             // 0..31
            int nc = (t & 7) * 16;       // 0..112
            const _Float16* src = &Bm[(long)(k0 + kr) * N + n0 + nc];
            v8h b0 = *(const v8h*)src;
            v8h b1 = *(const v8h*)(src + 8);
            #pragma unroll
            for (int j = 0; j < 8; ++j) {
                Bsr[nc + j][kr]     = b0[j];
                Bsr[nc + 8 + j][kr] = b1[j];
            }
        }
#ifdef HAVE_ASYNC_LDS
        wait_async0();
#endif
        __syncthreads();
        V16H A, Bf;
        {
            int row = lane & 15;
            int kh  = (lane < 16) ? 0 : 8;
            A.h[0] = *(const v8h*)&Asr[row][kh];
            A.h[1] = *(const v8h*)&Asr[row][kh + 16];
            int nrow = wv * 16 + (lane & 15);
            int kb   = (lane < 16) ? 0 : 8;
            Bf.h[0] = *(const v8h*)&Bsr[nrow][kb];
            Bf.h[1] = *(const v8h*)&Bsr[nrow][kb + 16];
        }
        acc = __builtin_amdgcn_wmma_f32_16x16x32_f16(false, A.v, false, Bf.v, (short)0, acc, false, false);
        __syncthreads();
    }
    #pragma unroll
    for (int r = 0; r < 8; ++r) {
        int row = (lane < 16) ? r : r + 8;
        int col = n0 + wv * 16 + (lane & 15);
        if (row < M && col < N) {
            float v = acc[r] + bias[col];
            if (relu) v = v > 0.f ? v : 0.f;
            Cm[(long)row * N + col] = (_Float16)v;
        }
    }
}

// ---------------------------------------------------------------- heads (tiny GEMV)
__global__ void heads_k(const _Float16* __restrict__ lat,  // [8][256]
    const float* wg, const float* bg, const float* ww, const float* bw,
    const float* wga, const float* bga, const float* wsh, const float* bsh,
    const float* wd, const float* bd, const float* wc, const float* bc,
    const float* wt, const float* bt,
    float* __restrict__ raw)                               // [8][22]
{
    int t = blockIdx.x * blockDim.x + threadIdx.x;
    if (t >= 8 * 22) return;
    int b = t / 22, d = t % 22;
    const float* w; const float* bi; int col, dim;
    if (d < 7)       { w = wg;  bi = bg;  col = d;      dim = 7; }
    else if (d < 10) { w = ww;  bi = bw;  col = d - 7;  dim = 3; }
    else if (d == 10){ w = wga; bi = bga; col = 0;      dim = 1; }
    else if (d == 11){ w = wsh; bi = bsh; col = 0;      dim = 1; }
    else if (d == 12){ w = wd;  bi = bd;  col = 0;      dim = 1; }
    else if (d == 13){ w = wc;  bi = bc;  col = 0;      dim = 1; }
    else             { w = wt;  bi = bt;  col = d - 14; dim = 8; }
    float s = bi[col];
    for (int k = 0; k < 256; ++k) s += (float)lat[b * 256 + k] * w[k * dim + col];
    raw[t] = s;
}

// pp layout per batch (23 floats): [0..6] gate, [7..9] wbn, [10] gamma, [11] sharp y,
// [12] omega, [13] alpha, [14..21] tc, [22] tcsum
__global__ void head_act_k(const float* __restrict__ raw, float* __restrict__ pp,
                           float* __restrict__ gate_out)
{
    int b = threadIdx.x;
    if (b >= 8) return;
    const float* r = raw + b * 22;
    float* P = pp + b * 23;
    for (int i = 0; i < 7; ++i) {
        float g = trangef(r[i], 0.01f, 1.0f);
        P[i] = g; gate_out[b * 7 + i] = g;
    }
    float w0 = expf(trangef(r[7], -0.5f, 0.5f));
    float w1 = expf(trangef(r[8], -0.5f, 0.5f));
    float w2 = expf(trangef(r[9], -0.5f, 0.5f));
    float cs = 1.0f / (1e-5f + 0.27f * w0 + 0.67f * w1 + 0.06f * w2);
    P[7] = cs * w0; P[8] = cs * w1; P[9] = cs * w2;
    const float lg = 0.9162907318741551f; // log 2.5
    P[10] = expf(trangef(r[10], -lg, lg));
    P[11] = trangef(r[11], 0.1f, 1.0f);
    P[12] = trangef(r[12], 0.1f, 1.0f);
    P[13] = tanhf(r[13]);
    float s = 0.f;
    for (int i = 0; i < 8; ++i) { float tc = trangef(r[14 + i], 0.5f, 2.0f); P[14 + i] = tc; s += tc; }
    P[22] = s + 1e-30f;
}

// ---------------------------------------------------------------- DIP tail
#define HWIMG 50176            // 224*224
#define NPX   1204224L         // 8*3*224*224

__global__ void init_k(unsigned* hist, float* sums, unsigned* mm) {
    int i = blockIdx.x * blockDim.x + threadIdx.x;
    if (i < 2048) hist[i] = 0u;
    if (i < 32)   sums[i] = 0.f;
    if (i < 14)   mm[i] = (i & 1) ? 0u : 0xFFFFFFFFu;  // [min,max] per buffer
}

__global__ void dark_k(const float* __restrict__ x, float* __restrict__ dark) {
    long i = blockIdx.x * (long)blockDim.x + threadIdx.x;
    if (i >= 8L * HWIMG) return;
    int b = (int)(i / HWIMG); int p = (int)(i % HWIMG);
    const float* px = x + (long)b * 3 * HWIMG + p;
    dark[i] = fminf(px[0], fminf(px[HWIMG], px[2 * HWIMG]));
}

__global__ void atm_hist_k(const float* __restrict__ dark, unsigned* __restrict__ hist) {
    long i = blockIdx.x * (long)blockDim.x + threadIdx.x;
    if (i >= 8L * HWIMG) return;
    int b = (int)(i / HWIMG);
    int bin = (int)(fminf(fmaxf(dark[i], 0.f), 0.999999f) * 256.f);
    atomicAdd(&hist[b * 256 + bin], 1u);
}

__global__ void atm_thresh_k(const unsigned* __restrict__ hist, float* __restrict__ thr) {
    int b = threadIdx.x; if (b >= 8) return;
    int acc = 0, bin = 255;
    for (; bin >= 0; --bin) { acc += (int)hist[b * 256 + bin]; if (acc >= 50) break; }
    if (bin < 0) bin = 0;
    thr[b] = bin / 256.f;
}

__global__ void atm_sum_k(const float* __restrict__ x, const float* __restrict__ dark,
                          const float* __restrict__ thr, float* __restrict__ sums) {
    long i = blockIdx.x * (long)blockDim.x + threadIdx.x;
    if (i >= 8L * HWIMG) return;
    int b = (int)(i / HWIMG); int p = (int)(i % HWIMG);
    if (dark[i] >= thr[b]) {
        atomicAdd(&sums[b * 4 + 0], x[((long)b * 3 + 0) * HWIMG + p]);
        atomicAdd(&sums[b * 4 + 1], x[((long)b * 3 + 1) * HWIMG + p]);
        atomicAdd(&sums[b * 4 + 2], x[((long)b * 3 + 2) * HWIMG + p]);
        atomicAdd(&sums[b * 4 + 3], 1.f);
    }
}

__global__ void atm_final_k(const float* __restrict__ sums, float* __restrict__ afog) {
    int b = threadIdx.x; if (b >= 8) return;
    float c = fmaxf(sums[b * 4 + 3], 1.f);
    for (int k = 0; k < 3; ++k) afog[b * 3 + k] = fmaxf(sums[b * 4 + k] / c, 1e-6f);
}

DEV int refl224(int i) { return i < 0 ? -i : (i > 223 ? 446 - i : i); }

__global__ void blur_h_k(const float* __restrict__ x, float* __restrict__ tmp) {
    long i = blockIdx.x * (long)blockDim.x + threadIdx.x;
    if (i >= NPX) return;
    int xx = (int)(i % 224); long r = i / 224;
    int yy = (int)(r % 224); long bc = r / 224;
    float g[13]; float s = 0.f;
    #pragma unroll
    for (int j = 0; j < 13; ++j) { float c = (j - 6) * (1.f / 2.55f); g[j] = __expf(-0.5f * c * c); s += g[j]; }
    float acc = 0.f;
    #pragma unroll
    for (int j = 0; j < 13; ++j)
        acc += g[j] * x[(bc * 224 + yy) * 224 + refl224(xx + j - 6)];
    tmp[i] = acc / s;
}

__global__ void blur_v_k(const float* __restrict__ tmp, float* __restrict__ out) {
    long i = blockIdx.x * (long)blockDim.x + threadIdx.x;
    if (i >= NPX) return;
    int xx = (int)(i % 224); long r = i / 224;
    int yy = (int)(r % 224); long bc = r / 224;
    float g[13]; float s = 0.f;
    #pragma unroll
    for (int j = 0; j < 13; ++j) { float c = (j - 6) * (1.f / 2.55f); g[j] = __expf(-0.5f * c * c); s += g[j]; }
    float acc = 0.f;
    #pragma unroll
    for (int j = 0; j < 13; ++j)
        acc += g[j] * tmp[(bc * 224 + refl224(yy + j - 6)) * 224 + xx];
    out[i] = acc / s;
}

__global__ void dip_main_k(const float* __restrict__ x, const float* __restrict__ blur,
                           const float* __restrict__ afog, const float* __restrict__ pp,
                           float* __restrict__ pre)   // [6][NPX]
{
    long i = blockIdx.x * (long)blockDim.x + threadIdx.x;
    if (i >= 8L * HWIMG) return;
    int b = (int)(i / HWIMG); int p = (int)(i % HWIMG);
    const float* P = pp + b * 23;
    long base = (long)b * 3 * HWIMG + p;
    float ch[3] = { x[base], x[base + HWIMG], x[base + 2 * HWIMG] };
    float aa[3] = { afog[b * 3], afog[b * 3 + 1], afog[b * 3 + 2] };
    float idark = fminf(ch[0] / aa[0], fminf(ch[1] / aa[1], ch[2] / aa[2]));
    float tfog = fmaxf(1.f - P[12] * idark, 0.01f);
    float lum = fminf(fmaxf(0.27f * ch[0] + 0.67f * ch[1] + 0.06f * ch[2], 0.f), 1.f);
    float clum = -__cosf(3.14159265358979f * lum) * 0.5f + 0.5f;
    float al = P[13];
    #pragma unroll
    for (int c = 0; c < 3; ++c) {
        long o = base + (long)c * HWIMG;
        float xv = ch[c];
        pre[0 * NPX + o] = P[7 + c] * xv;                                  // wb
        pre[1 * NPX + o] = __powf(fmaxf(xv, 1e-4f), P[10]);                // gamma
        pre[2 * NPX + o] = xv + P[11] * (xv - blur[o]);                    // sharp
        pre[3 * NPX + o] = (xv - aa[c]) / tfog + aa[c];                    // defog
        float cimg = xv / (lum + 1e-6f) * clum;                            // contrast
        pre[4 * NPX + o] = (1.f - al) * xv + al * cimg;
        float tot = 0.f;                                                   // tone
        #pragma unroll
        for (int s8 = 0; s8 < 8; ++s8)
            tot += fminf(fmaxf(xv - s8 * 0.125f, 0.f), 0.125f) * P[14 + s8];
        pre[5 * NPX + o] = tot * (8.f / P[22]);
    }
}

__global__ void minmax_k(const float* __restrict__ data, long npx, int nbuf,
                         unsigned* __restrict__ mm)   // [nbuf][2]
{
    __shared__ unsigned smn[256], smx[256];
    long i = blockIdx.x * (long)blockDim.x + threadIdx.x;
    long total = npx * nbuf;
    unsigned mn = 0xFFFFFFFFu, mx = 0u; int bufi = 0;
    if (i < total) { bufi = (int)(i / npx); unsigned e = f2ord(data[i]); mn = e; mx = e; }
    smn[threadIdx.x] = mn; smx[threadIdx.x] = mx;
    __syncthreads();
    for (int s = 128; s > 0; s >>= 1) {
        if ((int)threadIdx.x < s) {
            smn[threadIdx.x] = min(smn[threadIdx.x], smn[threadIdx.x + s]);
            smx[threadIdx.x] = max(smx[threadIdx.x], smx[threadIdx.x + s]);
        }
        __syncthreads();
    }
    if (threadIdx.x == 0 && i < total) {
        atomicMin(&mm[bufi * 2],     smn[0]);
        atomicMax(&mm[bufi * 2 + 1], smx[0]);
    }
}

__global__ void combine_k(const float* __restrict__ pre, const unsigned* __restrict__ mm,
                          const float* __restrict__ pp, float* __restrict__ comb)
{
    long i = blockIdx.x * (long)blockDim.x + threadIdx.x;
    if (i >= NPX) return;
    int b = (int)(i / (3 * HWIMG));
    const float* P = pp + b * 23;
    float s = 0.f;
    #pragma unroll
    for (int k = 0; k < 6; ++k) {
        float mn = ord2f(mm[k * 2]), mx = ord2f(mm[k * 2 + 1]);
        s += (pre[k * NPX + i] - mn) / (mx - mn) * P[k];
    }
    comb[i] = s;
}

__global__ void final_k(const float* __restrict__ comb, const unsigned* __restrict__ mm6,
                        float* __restrict__ out)
{
    long i = blockIdx.x * (long)blockDim.x + threadIdx.x;
    if (i >= NPX) return;
    float mn = ord2f(mm6[0]), mx = ord2f(mm6[1]);
    out[i] = (comb[i] - mn) / (mx - mn);
}

// ---------------------------------------------------------------- host
extern "C" void kernel_launch(void* const* d_in, const int* in_sizes, int n_in,
                              void* d_out, int out_size, void* d_ws, size_t ws_size,
                              hipStream_t stream)
{
    (void)in_sizes; (void)n_in; (void)out_size; (void)ws_size;
    const float* x_f32 = (const float*)d_in[0];
    // input index map: convs w/b at 1+2i / 2+2i (i=0..12), fcs at 27+2i / 28+2i, heads from 33
    struct LCfg { int cin, cout, hw, pool; };
    static const LCfg L[13] = {
        {3,64,224,0},{64,64,224,1},{64,128,112,0},{128,128,112,1},
        {128,256,56,0},{256,256,56,0},{256,256,56,1},
        {256,512,28,0},{512,512,28,0},{512,512,28,1},
        {512,512,14,0},{512,512,14,0},{512,512,14,1}
    };

    char* wsb = (char*)d_ws;
    size_t off = 0;
    auto carve = [&](size_t bytes) -> void* {
        void* p = wsb + off;
        off = (off + bytes + 255) & ~(size_t)255;
        return p;
    };
    const size_t ACT = 25690112;                 // max f16 activation elems (8*64*224*224)
    _Float16* bufA = (_Float16*)carve(ACT * 2);
    _Float16* bufB = (_Float16*)carve(ACT * 2);
    size_t convWoff[13], convWtot = 0;
    for (int i = 0; i < 13; ++i) { convWoff[i] = convWtot; convWtot += (size_t)L[i].cout * L[i].cin * 9; }
    _Float16* convW = (_Float16*)carve(convWtot * 2);
    const size_t fcWsz[3] = { 25088UL * 4096, 4096UL * 4096, 4096UL * 256 };
    _Float16* fcW[3];
    for (int i = 0; i < 3; ++i) fcW[i] = (_Float16*)carve(fcWsz[i] * 2);
    float*    raw   = (float*)carve(8 * 22 * 4);
    float*    pp    = (float*)carve(8 * 23 * 4);
    float*    dark  = (float*)carve(8L * HWIMG * 4);
    unsigned* hist  = (unsigned*)carve(2048 * 4);
    float*    sums  = (float*)carve(32 * 4);
    float*    afog  = (float*)carve(24 * 4);
    float*    thr   = (float*)carve(8 * 4);
    float*    blurT = (float*)carve(NPX * 4);
    float*    blurO = (float*)carve(NPX * 4);
    float*    pre   = (float*)carve(6 * NPX * 4);
    float*    comb  = (float*)carve(NPX * 4);
    unsigned* mm    = (unsigned*)carve(14 * 4);

    const int TB = 256;
    auto blocks = [](long n, int tb) { return (unsigned)((n + tb - 1) / tb); };

    // --- convert input + weights to f16
    f32_to_f16_k<<<blocks(NPX, TB), TB, 0, stream>>>(x_f32, bufA, NPX);
    for (int i = 0; i < 13; ++i) {
        long n = (long)L[i].cout * L[i].cin * 9;
        f32_to_f16_k<<<blocks(n, TB), TB, 0, stream>>>((const float*)d_in[1 + 2 * i], convW + convWoff[i], n);
    }
    for (int i = 0; i < 3; ++i)
        f32_to_f16_k<<<blocks((long)fcWsz[i], TB), TB, 0, stream>>>((const float*)d_in[27 + 2 * i], fcW[i], (long)fcWsz[i]);

    // --- VGG16 convs
    _Float16* cur = bufA; _Float16* oth = bufB;
    for (int i = 0; i < 13; ++i) {
        int H = L[i].hw, W = L[i].hw, Cout = L[i].cout, Cin = L[i].cin;
        long N = 8L * H * W;
        dim3 grid(blocks(N, 64), (Cout + 63) / 64);
        conv3x3_wmma_k<<<grid, 256, 0, stream>>>(cur, convW + convWoff[i],
                                                 (const float*)d_in[2 + 2 * i], oth,
                                                 Cin, Cout, H, W);
        if (L[i].pool) {
            long np = 8L * Cout * (H / 2) * (W / 2);
            maxpool2_k<<<blocks(np, TB), TB, 0, stream>>>(oth, cur, Cout, H, W);
            // cur now holds pooled output
        } else {
            _Float16* t = cur; cur = oth; oth = t;
        }
    }
    // cur = pool5 = z : [8][25088] f16

    // --- FC layers (all followed by ReLU; fc3's ReLU is latent = relu(vgg16))
    gemm_wmma_k<<<dim3(4096 / 128), 256, 0, stream>>>(cur, fcW[0], (const float*)d_in[28], oth, 8, 4096, 25088, 1);
    { _Float16* t = cur; cur = oth; oth = t; }
    gemm_wmma_k<<<dim3(4096 / 128), 256, 0, stream>>>(cur, fcW[1], (const float*)d_in[30], oth, 8, 4096, 4096, 1);
    { _Float16* t = cur; cur = oth; oth = t; }
    gemm_wmma_k<<<dim3(2), 256, 0, stream>>>(cur, fcW[2], (const float*)d_in[32], oth, 8, 256, 4096, 1);
    const _Float16* lat = oth;   // [8][256]

    // --- heads
    float* gate_out = (float*)d_out + NPX;
    heads_k<<<1, 256, 0, stream>>>(lat,
        (const float*)d_in[33], (const float*)d_in[34], (const float*)d_in[35], (const float*)d_in[36],
        (const float*)d_in[37], (const float*)d_in[38], (const float*)d_in[39], (const float*)d_in[40],
        (const float*)d_in[41], (const float*)d_in[42], (const float*)d_in[43], (const float*)d_in[44],
        (const float*)d_in[45], (const float*)d_in[46], raw);
    head_act_k<<<1, 32, 0, stream>>>(raw, pp, gate_out);

    // --- DIP tail
    init_k<<<8, 256, 0, stream>>>(hist, sums, mm);
    dark_k<<<blocks(8L * HWIMG, TB), TB, 0, stream>>>(x_f32, dark);
    atm_hist_k<<<blocks(8L * HWIMG, TB), TB, 0, stream>>>(dark, hist);
    atm_thresh_k<<<1, 32, 0, stream>>>(hist, thr);
    atm_sum_k<<<blocks(8L * HWIMG, TB), TB, 0, stream>>>(x_f32, dark, thr, sums);
    atm_final_k<<<1, 32, 0, stream>>>(sums, afog);
    blur_h_k<<<blocks(NPX, TB), TB, 0, stream>>>(x_f32, blurT);
    blur_v_k<<<blocks(NPX, TB), TB, 0, stream>>>(blurT, blurO);
    dip_main_k<<<blocks(8L * HWIMG, TB), TB, 0, stream>>>(x_f32, blurO, afog, pp, pre);
    minmax_k<<<blocks(6 * NPX, TB), TB, 0, stream>>>(pre, NPX, 6, mm);
    combine_k<<<blocks(NPX, TB), TB, 0, stream>>>(pre, mm, pp, comb);
    minmax_k<<<blocks(NPX, TB), TB, 0, stream>>>(comb, NPX, 1, mm + 12);
    final_k<<<blocks(NPX, TB), TB, 0, stream>>>(comb, mm + 12, (float*)d_out);
}
